// Pose_Encoder_77137612636341
// MI455X (gfx1250) — matrix operand order repaired and checked by reading, not compiled, not run on hardware
//
#include <hip/hip_runtime.h>
#include <hip/hip_bf16.h>
#include <math.h>

// ---------------- sizes ----------------
#define B_  32
#define T_  64
#define G_  (B_ * T_)          // 2048
#define HID_ 32
#define NH_  4
#define HD_  128
#define R_   256
#define S_   65                // T+1 tokens
#define NROW (B_ * S_)         // 2080

typedef __attribute__((ext_vector_type(16))) _Float16 v16h;
typedef __attribute__((ext_vector_type(8)))  float    v8f;

// ---------------- weight pre-pack into WMMA B-fragment layout ----------------
// Logical B is [K,N]. TRANSB==0: Bm is f32 [K,N] row-major.
//                     TRANSB==1: Bm is f32 [N,K] row-major (B[k][n]=Bm[n*K+k]).
// Output halves: out[((tn*(K/32)+kb)*32 + lane)*16 + i] = B[kb*32+(lane>>4)*16+i][tn*16+(lane&15)]
template <int TRANSB>
__global__ void pack_b(const float* __restrict__ Bm, _Float16* __restrict__ out,
                       int N, int K)
{
    int idx = blockIdx.x * blockDim.x + threadIdx.x;
    if (idx >= N * K) return;
    const int KB = K >> 5;
    int i    = idx & 15;
    int lane = (idx >> 4) & 31;
    int blk  = idx >> 9;              // tn*KB + kb
    int tn = blk / KB, kb = blk - tn * KB;
    int n = tn * 16 + (lane & 15);
    int k = kb * 32 + (lane >> 4) * 16 + i;
    float v = TRANSB ? Bm[(size_t)n * K + k] : Bm[(size_t)k * N + n];
    out[idx] = (_Float16)v;
}

// ---------------- WMMA GEMM: C[M,N] = act(A[M,K] x B + bias) ----------------
// A: f32 row-major [M,K] (K mult of 32, rows 16B aligned).
// Bp: pre-packed f16 fragments (see pack_b). N mult of 64, M mult of 16.
// Block = 128 threads = 4 waves; wave w -> M-tile (blockIdx.x*4+w), covering a
// 16x64 output strip with 4 accumulators (A fragment shared by 4 WMMAs).
// grid = (ceil(M/64), N/64).
template <int ACT>
__global__ __launch_bounds__(128)
void gemm_wmma(const float* __restrict__ A, const _Float16* __restrict__ Bp,
               const float* __restrict__ bias, float* __restrict__ C,
               int M, int N, int K)
{
    const int lane  = threadIdx.x & 31;
    const int wave  = threadIdx.x >> 5;
    const int tm    = blockIdx.x * 4 + wave;   // M tile index
    if (tm * 16 >= M) return;                  // wave-uniform: EXEC stays full
    const int tn0   = blockIdx.y * 4;          // first of 4 N tiles
    const int group = lane >> 4;               // 0: low K half, 1: high K half
    const int row   = tm * 16 + (lane & 15);   // A row for this lane
    const int ncol  = lane & 15;               // within-tile column
    const int KB    = K >> 5;

    const float4* Arow = (const float4*)(A + (size_t)row * K);
    const v16h*   Bv   = (const v16h*)Bp;

    v8f acc0 = {}, acc1 = {}, acc2 = {}, acc3 = {};
    for (int kb = 0; kb < KB; ++kb) {
        const int kk = kb * 32;
        if (kb + 1 < KB) {
            __builtin_prefetch(&A[(size_t)row * K + kk + 32], 0, 0);
        }
        // A 16x32 f16 layout: lane l (l<16): M=l, halves = K{0..15}
        //                     lane l>=16 : M=l-16, halves = K{16..31}
        float4 a0 = Arow[(kk + group * 16) >> 2];
        float4 a1 = Arow[((kk + group * 16) >> 2) + 1];
        float4 a2 = Arow[((kk + group * 16) >> 2) + 2];
        float4 a3 = Arow[((kk + group * 16) >> 2) + 3];
        v16h a;
        a[0] = (_Float16)a0.x; a[1] = (_Float16)a0.y; a[2] = (_Float16)a0.z; a[3] = (_Float16)a0.w;
        a[4] = (_Float16)a1.x; a[5] = (_Float16)a1.y; a[6] = (_Float16)a1.z; a[7] = (_Float16)a1.w;
        a[8]  = (_Float16)a2.x; a[9]  = (_Float16)a2.y; a[10] = (_Float16)a2.z; a[11] = (_Float16)a2.w;
        a[12] = (_Float16)a3.x; a[13] = (_Float16)a3.y; a[14] = (_Float16)a3.z; a[15] = (_Float16)a3.w;
        // B fragments: one aligned 32-byte load per tile
        v16h b0 = Bv[((size_t)(tn0 + 0) * KB + kb) * 32 + lane];
        v16h b1 = Bv[((size_t)(tn0 + 1) * KB + kb) * 32 + lane];
        v16h b2 = Bv[((size_t)(tn0 + 2) * KB + kb) * 32 + lane];
        v16h b3 = Bv[((size_t)(tn0 + 3) * KB + kb) * 32 + lane];
        acc0 = __builtin_amdgcn_wmma_f32_16x16x32_f16(false, a, false, b0, (short)0, acc0, false, false);
        acc1 = __builtin_amdgcn_wmma_f32_16x16x32_f16(false, a, false, b1, (short)0, acc1, false, false);
        acc2 = __builtin_amdgcn_wmma_f32_16x16x32_f16(false, a, false, b2, (short)0, acc2, false, false);
        acc3 = __builtin_amdgcn_wmma_f32_16x16x32_f16(false, a, false, b3, (short)0, acc3, false, false);
    }
    // C/D layout: VGPR r holds (M = tm*16 + r + 8*group, N = tile col)
#pragma unroll
    for (int t = 0; t < 4; ++t) {
        const v8f& acc = (t == 0) ? acc0 : (t == 1) ? acc1 : (t == 2) ? acc2 : acc3;
        int col = (tn0 + t) * 16 + ncol;
        float bv = bias ? bias[col] : 0.0f;
#pragma unroll
        for (int r = 0; r < 8; ++r) {
            int m = tm * 16 + r + group * 8;
            float v = acc[r] + bv;
            if (ACT) v = fmaxf(v, 0.0f);
            C[(size_t)m * N + col] = v;
        }
    }
}

// ---------------- GAT layer 0 linear (d_in = 3) ----------------
__global__ void linear3(const float* __restrict__ x, const float* __restrict__ W,
                        float* __restrict__ h, int Ntot)
{
    int idx = blockIdx.x * blockDim.x + threadIdx.x;
    if (idx >= Ntot * HD_) return;
    int n = idx >> 7, c = idx & 127;
    const float* xp = x + (size_t)n * 3;
    h[idx] = xp[0] * W[c] + xp[1] * W[HD_ + c] + xp[2] * W[2 * HD_ + c];
}

// ---------------- attention coefficients es/ed ----------------
__global__ void gat_coef(const float* __restrict__ h,
                         const float* __restrict__ a_s, const float* __restrict__ a_d,
                         float* __restrict__ es, float* __restrict__ ed, int Ntot)
{
    int idx = blockIdx.x * blockDim.x + threadIdx.x;   // n*4 + head
    if (idx >= Ntot * NH_) return;
    int n = idx >> 2, hh = idx & 3;
    const float* hp = h + (size_t)n * HD_ + hh * HID_;
    const float* as = a_s + hh * HID_;
    const float* ad = a_d + hh * HID_;
    float s = 0.f, d = 0.f;
#pragma unroll
    for (int c = 0; c < HID_; ++c) { float v = hp[c]; s += v * as[c]; d += v * ad[c]; }
    es[idx] = s; ed[idx] = d;
}

// ---------------- per-graph GAT aggregation + LN + ELU ----------------
#define MAXN 68
#define MAXE 256   // max edges/graph = 134 directed + 68 loops = 202

__device__ __forceinline__ unsigned fenc(float x) {
    unsigned b = __float_as_uint(x);
    return (b & 0x80000000u) ? ~b : (b | 0x80000000u);
}
__device__ __forceinline__ float fdec(unsigned u) {
    return (u & 0x80000000u) ? __uint_as_float(u ^ 0x80000000u) : __uint_as_float(~u);
}

__global__ __launch_bounds__(256)
void gat_agg(const float* __restrict__ h, const float* __restrict__ es,
             const float* __restrict__ ed,
             const int* __restrict__ src, const int* __restrict__ dst,
             const float* __restrict__ bias, const float* __restrict__ gamma,
             const float* __restrict__ beta,
             float* __restrict__ out, int n_pts, int Ed)
{
    __shared__ _Float16 sm_h[MAXN * HD_];     // 17408 B
    __shared__ float    sm_out[MAXN * HD_];   // 34816 B
    __shared__ float    sm_es[MAXN * NH_], sm_ed[MAXN * NH_];
    __shared__ unsigned sm_m[MAXN * NH_];
    __shared__ float    sm_den[MAXN * NH_];
    __shared__ float    sm_l[MAXE * NH_];
    __shared__ short    sm_s[MAXE], sm_d[MAXE];

    const int g = blockIdx.x, tid = threadIdx.x;
    const int E = Ed + n_pts;
    const int base = g * n_pts;

    for (int i = tid; i < n_pts * HD_; i += 256) {
        sm_h[i]   = (_Float16)h[(size_t)base * HD_ + i];
        sm_out[i] = 0.f;
    }
    for (int i = tid; i < n_pts * NH_; i += 256) {
        sm_es[i] = es[base * NH_ + i];
        sm_ed[i] = ed[base * NH_ + i];
        sm_m[i]  = fenc(-1e30f);
        sm_den[i] = 0.f;
    }
    for (int j = tid; j < E; j += 256) {
        int s, d;
        if (j < Ed) { s = src[(size_t)g * Ed + j] - base; d = dst[(size_t)g * Ed + j] - base; }
        else        { s = d = j - Ed; }   // self loops
        sm_s[j] = (short)s; sm_d[j] = (short)d;
    }
    __syncthreads();

    // pass 1: leaky-relu logits + segment max (order-independent)
    for (int j = tid; j < E; j += 256) {
        int s = sm_s[j], d = sm_d[j];
#pragma unroll
        for (int hh = 0; hh < NH_; ++hh) {
            float l = sm_es[s * NH_ + hh] + sm_ed[d * NH_ + hh];
            l = (l > 0.f) ? l : 0.2f * l;
            sm_l[j * NH_ + hh] = l;
            atomicMax(&sm_m[d * NH_ + hh], fenc(l));
        }
    }
    __syncthreads();

    // pass 2: exp + segment sum
    for (int j = tid; j < E; j += 256) {
        int d = sm_d[j];
#pragma unroll
        for (int hh = 0; hh < NH_; ++hh) {
            float m = fdec(sm_m[d * NH_ + hh]);
            float ex = expf(sm_l[j * NH_ + hh] - m);
            sm_l[j * NH_ + hh] = ex;
            atomicAdd(&sm_den[d * NH_ + hh], ex);
        }
    }
    __syncthreads();

    // pass 3: weighted aggregation of h[src] into out[dst]
    for (int idx = tid; idx < E * HD_; idx += 256) {
        int j = idx >> 7, c = idx & 127, hh = c >> 5;
        int s = sm_s[j], d = sm_d[j];
        float alpha = sm_l[j * NH_ + hh] / (sm_den[d * NH_ + hh] + 1e-16f);
        atomicAdd(&sm_out[d * HD_ + c], (float)sm_h[s * HD_ + c] * alpha);
    }
    __syncthreads();

    // bias + LayerNorm + ELU, write back
    for (int n = tid; n < n_pts; n += 256) {
        float mu = 0.f;
        for (int c = 0; c < HD_; ++c) mu += sm_out[n * HD_ + c] + bias[c];
        mu *= (1.0f / HD_);
        float var = 0.f;
        for (int c = 0; c < HD_; ++c) {
            float v = sm_out[n * HD_ + c] + bias[c] - mu;
            var += v * v;
        }
        var *= (1.0f / HD_);
        float rs = rsqrtf(var + 1e-5f);
        for (int c = 0; c < HD_; ++c) {
            float v = (sm_out[n * HD_ + c] + bias[c] - mu) * rs * gamma[c] + beta[c];
            v = (v > 0.f) ? v : (expf(v) - 1.0f);
            out[(size_t)(base + n) * HD_ + c] = v;
        }
    }
}

// ---------------- mean pool over nodes -> concat buffer ----------------
__global__ void pool_mean(const float* __restrict__ x, float* __restrict__ cat,
                          int n_pts, int off)
{
    int idx = blockIdx.x * blockDim.x + threadIdx.x;  // g*128 + c
    if (idx >= G_ * HD_) return;
    int g = idx >> 7, c = idx & 127;
    float s = 0.f;
    for (int i = 0; i < n_pts; ++i) s += x[(size_t)(g * n_pts + i) * HD_ + c];
    cat[(size_t)g * (3 * HD_) + off + c] = s / (float)n_pts;
}

// ---------------- token assembly: cls + pos ----------------
__global__ void build_tok(const float* __restrict__ xproj, const float* __restrict__ cls,
                          const float* __restrict__ pos, float* __restrict__ tok)
{
    int idx = blockIdx.x * blockDim.x + threadIdx.x;  // row*256 + c
    if (idx >= NROW * R_) return;
    int row = idx >> 8, c = idx & 255;
    int b = row / S_, s = row % S_;
    float v = (s == 0) ? cls[c] : xproj[(size_t)(b * T_ + s - 1) * R_ + c];
    tok[idx] = v + pos[s * R_ + c];
}

// ---------------- scalar-head attention (head_dim = 1, scale = 1) ----------------
__global__ __launch_bounds__(96)
void attn_scalar(const float* __restrict__ qkv, float* __restrict__ out)
{
    const int b = blockIdx.x, hh = blockIdx.y;   // 32 x 256
    __shared__ float kk[S_], vv[S_];
    int tid = threadIdx.x;
    if (tid < S_) {
        size_t r = (size_t)(b * S_ + tid) * (3 * R_);
        kk[tid] = qkv[r + R_ + hh];
        vv[tid] = qkv[r + 2 * R_ + hh];
    }
    __syncthreads();
    if (tid < S_) {
        float q = qkv[(size_t)(b * S_ + tid) * (3 * R_) + hh];
        float m = -1e30f;
        for (int j = 0; j < S_; ++j) m = fmaxf(m, q * kk[j]);
        float den = 0.f, o = 0.f;
        for (int j = 0; j < S_; ++j) {
            float e = expf(q * kk[j] - m);
            den += e; o += e * vv[j];
        }
        out[(size_t)(b * S_ + tid) * R_ + hh] = o / den;
    }
}

// ---------------- residual add + LayerNorm (row length 256) ----------------
__global__ __launch_bounds__(256)
void res_ln(float* __restrict__ tok, const float* __restrict__ add,
            const float* __restrict__ gamma, const float* __restrict__ beta)
{
    __shared__ float red[R_];
    int row = blockIdx.x, c = threadIdx.x;
    size_t o = (size_t)row * R_ + c;
    float v = tok[o] + add[o];
    red[c] = v; __syncthreads();
    for (int s = 128; s > 0; s >>= 1) { if (c < s) red[c] += red[c + s]; __syncthreads(); }
    float mu = red[0] * (1.0f / R_); __syncthreads();
    float d = v - mu;
    red[c] = d * d; __syncthreads();
    for (int s = 128; s > 0; s >>= 1) { if (c < s) red[c] += red[c + s]; __syncthreads(); }
    float var = red[0] * (1.0f / R_);
    tok[o] = d * rsqrtf(var + 1e-5f) * gamma[c] + beta[c];
}

// ---------------- CLS extraction ----------------
__global__ void take_cls(const float* __restrict__ tok, float* __restrict__ out)
{
    int idx = blockIdx.x * blockDim.x + threadIdx.x;
    if (idx >= B_ * R_) return;
    int b = idx >> 8, c = idx & 255;
    out[idx] = tok[(size_t)b * S_ * R_ + c];
}

// =====================================================================
extern "C" void kernel_launch(void* const* d_in, const int* in_sizes, int n_in,
                              void* d_out, int out_size, void* d_ws, size_t ws_size,
                              hipStream_t stream)
{
    (void)in_sizes; (void)n_in; (void)out_size; (void)ws_size;
    auto F = [&](int i) { return (const float*)d_in[i]; };
    auto I = [&](int i) { return (const int*)d_in[i]; };

    // ---- workspace layout (floats; every offset is a multiple of 8 floats) ----
    const size_t NMAX = (size_t)G_ * MAXN;   // 139264
    float* ws    = (float*)d_ws;
    float* buf_x = ws;                         // NMAX*128
    float* buf_h = buf_x + NMAX * HD_;         // NMAX*128
    float* buf_es = buf_h + NMAX * HD_;        // NMAX*4
    float* buf_ed = buf_es + NMAX * NH_;       // NMAX*4
    float* cat   = buf_ed + NMAX * NH_;        // 2048*384
    float* xproj = cat + (size_t)G_ * 3 * HD_; // 2048*256
    float* tok   = xproj + (size_t)G_ * R_;    // 2080*256
    float* qkv   = tok + (size_t)NROW * R_;    // 2080*768
    float* attnb = qkv + (size_t)NROW * 3 * R_;// 2080*256
    float* tmpb  = attnb + (size_t)NROW * R_;  // 2080*256
    float* ffb   = tmpb + (size_t)NROW * R_;   // 2080*512
    _Float16* bpack = (_Float16*)(ffb + (size_t)NROW * 2 * R_); // max 768*256 halves

    auto mtiles = [](int M) { return (M / 16 + 3) / 4; };  // grid.x = ceil(M/64)

    // pack weight + run GEMM (weights are small; pack cost is negligible)
    auto run_gemm = [&](const float* A, const float* W, int transB,
                        const float* bias, float* Cout, int M, int N, int K, int act) {
        int tot = N * K;
        if (transB) pack_b<1><<<(tot + 255) / 256, 256, 0, stream>>>(W, bpack, N, K);
        else        pack_b<0><<<(tot + 255) / 256, 256, 0, stream>>>(W, bpack, N, K);
        dim3 gr(mtiles(M), N / 64);
        if (act) gemm_wmma<1><<<gr, 128, 0, stream>>>(A, bpack, bias, Cout, M, N, K);
        else     gemm_wmma<0><<<gr, 128, 0, stream>>>(A, bpack, bias, Cout, M, N, K);
    };

    // ---- GAT branches ----
    struct PartCfg { int n_pts, Ed, xi, si, di, pb, catoff; };
    const PartCfg parts[3] = {
        {17,  38, 0, 67, 68, 3,  0},     // body
        {68, 134, 1, 69, 70, 15, 128},   // head
        {42,  80, 2, 71, 72, 27, 256},   // hand
    };

    for (int p = 0; p < 3; ++p) {
        const PartCfg& pc = parts[p];
        const int Ntot = G_ * pc.n_pts;
        for (int l = 0; l < 2; ++l) {
            const int pb = pc.pb + l * 6;
            const float* W   = F(pb + 0);
            const float* a_s = F(pb + 1);
            const float* a_d = F(pb + 2);
            const float* bs  = F(pb + 3);
            const float* gm  = F(pb + 4);
            const float* bt  = F(pb + 5);
            if (l == 0) {
                linear3<<<(Ntot * HD_ + 255) / 256, 256, 0, stream>>>(
                    F(pc.xi), W, buf_h, Ntot);
            } else {
                run_gemm(buf_x, W, /*transB=*/0, nullptr, buf_h, Ntot, HD_, HD_, 0);
            }
            gat_coef<<<(Ntot * NH_ + 255) / 256, 256, 0, stream>>>(
                buf_h, a_s, a_d, buf_es, buf_ed, Ntot);
            gat_agg<<<G_, 256, 0, stream>>>(
                buf_h, buf_es, buf_ed, I(pc.si), I(pc.di),
                bs, gm, bt, buf_x, pc.n_pts, pc.Ed);
        }
        pool_mean<<<(G_ * HD_ + 255) / 256, 256, 0, stream>>>(
            buf_x, cat, pc.n_pts, pc.catoff);
    }

    // ---- projection: relu(cat @ proj_w + proj_b) ----
    run_gemm(cat, F(39), /*transB=*/0, F(40), xproj, G_, R_, 3 * HD_, 1);

    // ---- tokens ----
    build_tok<<<(NROW * R_ + 255) / 256, 256, 0, stream>>>(
        xproj, F(41), F(42), tok);

    // ---- transformer layers ----
    for (int l = 0; l < 2; ++l) {
        const int tb = 43 + l * 12;
        const float* in_w  = F(tb + 0);  const float* in_b  = F(tb + 1);
        const float* out_w = F(tb + 2);  const float* out_b = F(tb + 3);
        const float* g1 = F(tb + 4), *b1 = F(tb + 5);
        const float* g2 = F(tb + 6), *b2 = F(tb + 7);
        const float* w1 = F(tb + 8), *bb1 = F(tb + 9);
        const float* w2 = F(tb + 10), *bb2 = F(tb + 11);

        // qkv = tok @ in_w^T + in_b   (in_w is [768,256])
        run_gemm(tok, in_w, 1, in_b, qkv, NROW, 3 * R_, R_, 0);
        // per-head scalar attention
        attn_scalar<<<dim3(B_, R_), 96, 0, stream>>>(qkv, attnb);
        // out proj
        run_gemm(attnb, out_w, 1, out_b, tmpb, NROW, R_, R_, 0);
        res_ln<<<NROW, R_, 0, stream>>>(tok, tmpb, g1, b1);
        // feed-forward
        run_gemm(tok, w1, 1, bb1, ffb, NROW, 2 * R_, R_, 1);
        run_gemm(ffb, w2, 1, bb2, tmpb, NROW, R_, 2 * R_, 0);
        res_ln<<<NROW, R_, 0, stream>>>(tok, tmpb, g2, b2);
    }

    // ---- output: CLS tokens ----
    take_cls<<<(B_ * R_ + 255) / 256, 256, 0, stream>>>(tok, (float*)d_out);
}
